// MultiHeadAttention_50130858279236
// MI455X (gfx1250) — matrix operand order repaired and checked by reading, not compiled, not run on hardware
//
#include <hip/hip_runtime.h>

// MHA for MI455X (gfx1250): bf16 WMMA + fused flash-style attention that
// exploits CDNA5's 320KB-per-workgroup LDS (32x2048 fp32 score tile resident),
// async global->LDS staging and global prefetch where available.
// B=4, S=2048, D=1024, H=16, DK=64.

#define SLEN   2048
#define BATCH  4
#define NHEAD  16
#define DMODEL 1024
#define DKH    64
#define SROW   2052   // padded LDS row stride (floats) for the score tile

typedef __attribute__((ext_vector_type(16))) __bf16 v16bf;
typedef __attribute__((ext_vector_type(8)))  float  v8f;

#if defined(__has_builtin)
#if __has_builtin(__builtin_amdgcn_global_load_async_to_lds_b128)
#define HAVE_ASYNC_LDS 1
#endif
#endif

// Exact pointee type from the compiler diagnostic: GCC-style v4i vector.
typedef int a_v4i __attribute__((vector_size(16)));
typedef __attribute__((address_space(1))) a_v4i* gptr_v4i;   // global (AS1)
typedef __attribute__((address_space(3))) a_v4i* lptr_v4i;   // LDS (AS3)

__device__ __forceinline__ void wait_async_lds() {
#if defined(HAVE_ASYNC_LDS)
#if __has_builtin(__builtin_amdgcn_s_wait_asynccnt)
  __builtin_amdgcn_s_wait_asynccnt(0);
#else
  asm volatile("s_wait_asynccnt 0" ::: "memory");
#endif
#endif
}

union FragBF {
  v16bf        v;
  uint4        q[2];
  unsigned int u[8];
};

__device__ __forceinline__ unsigned short f2bf(float x) {
  unsigned int u = __float_as_uint(x);
  u += 0x7FFFu + ((u >> 16) & 1u);     // round-to-nearest-even
  return (unsigned short)(u >> 16);
}
__device__ __forceinline__ unsigned int pack2(float a, float b) {
  return (unsigned int)f2bf(a) | ((unsigned int)f2bf(b) << 16);
}

// ---------------------------------------------------------------------------
// Generic GEMM: C[M=8192, N=1024] = A[M,K=1024] * W[N,K]^T  (y = x @ W.T)
// Workgroup tile 128x64, 8 waves in 4x2, each wave 32x32 = 2x2 WMMA frags.
// ABF16:  A is bf16 (ctx; staged via async global->LDS DMA when available)
//         vs fp32 (inputs, converted to bf16 while staging)
// OUTMODE: 0 = fp32 row-major [M,N] (final output)
//          1 = bf16 [B,H,S,DK]   (Q, K)
//          2 = bf16 [B,H,DK,S]   (V transposed, k-contiguous for context)
// ---------------------------------------------------------------------------
template <bool ABF16, int OUTMODE>
__global__ __launch_bounds__(256)
void gemm_kernel(const void* __restrict__ Ain, const float* __restrict__ W,
                 void* __restrict__ Out) {
  __shared__ unsigned short As[128 * 32];
  __shared__ unsigned short Bs[64 * 32];

  const int tid = threadIdx.x;
  const int l   = tid & 31;
  const int w   = tid >> 5;
  const int wm  = w >> 1;          // 0..3
  const int wn  = w & 1;           // 0..1
  const int mbase = blockIdx.y * 128;
  const int nbase = blockIdx.x * 64;
  const int lm = l & 15;
  const int hi = l >> 4;

  v8f acc[2][2] = {};

  for (int k0 = 0; k0 < DMODEL; k0 += 32) {
    // ---- stage A tile 128x32 into LDS (bf16) ----
    {
      const int row = tid >> 1;
      const int kc  = (tid & 1) * 16;
      if (ABF16) {
        const uint4* src = (const uint4*)((const unsigned short*)Ain +
                            (size_t)(mbase + row) * DMODEL + k0 + kc);
        uint4* dst = (uint4*)&As[row * 32 + kc];
#if defined(HAVE_ASYNC_LDS)
        // Async DMA copy: no VGPR round-trip, tracked by ASYNCcnt.
        __builtin_amdgcn_global_load_async_to_lds_b128((gptr_v4i)src, (lptr_v4i)dst, 0, 0);
        __builtin_amdgcn_global_load_async_to_lds_b128((gptr_v4i)src, (lptr_v4i)dst, 16, 0);
#else
        dst[0] = src[0];
        dst[1] = src[1];
#endif
        if (k0 + 32 < DMODEL) __builtin_prefetch((const char*)(src) + 64, 0, 1);
      } else {
        const float4* src = (const float4*)((const float*)Ain +
                             (size_t)(mbase + row) * DMODEL + k0 + kc);
        unsigned int* dst = (unsigned int*)&As[row * 32 + kc];
#pragma unroll
        for (int i = 0; i < 4; ++i) {
          float4 x = src[i];
          dst[i * 2 + 0] = pack2(x.x, x.y);
          dst[i * 2 + 1] = pack2(x.z, x.w);
        }
        if (k0 + 32 < DMODEL) __builtin_prefetch((const char*)(src) + 128, 0, 1);
      }
    }
    // ---- stage B tile 64x32 (rows of W = N dim, K-contiguous) ----
    {
      const int row = tid >> 2;
      const int kc  = (tid & 3) * 8;
      const float4* src = (const float4*)(W + (size_t)(nbase + row) * DMODEL + k0 + kc);
      unsigned int* dst = (unsigned int*)&Bs[row * 32 + kc];
#pragma unroll
      for (int i = 0; i < 2; ++i) {
        float4 x = src[i];
        dst[i * 2 + 0] = pack2(x.x, x.y);
        dst[i * 2 + 1] = pack2(x.z, x.w);
      }
      if (k0 + 32 < DMODEL) __builtin_prefetch((const char*)(src) + 128, 0, 1);
    }
    if (ABF16) wait_async_lds();
    __syncthreads();

    FragBF a[2], b[2];
#pragma unroll
    for (int fm = 0; fm < 2; ++fm) {
      const unsigned short* rp = &As[(wm * 32 + fm * 16 + lm) * 32];
      a[fm].q[0] = *(const uint4*)(rp + hi * 8);        // K = hi*8 .. +7
      a[fm].q[1] = *(const uint4*)(rp + hi * 8 + 16);   // K = hi*8+16 .. +23
    }
#pragma unroll
    for (int fn = 0; fn < 2; ++fn) {
      const unsigned short* rp = &Bs[(wn * 32 + fn * 16 + lm) * 32 + hi * 8];
      b[fn].q[0] = ((const uint4*)rp)[0];
      b[fn].q[1] = ((const uint4*)rp)[2];
    }
#pragma unroll
    for (int fm = 0; fm < 2; ++fm)
#pragma unroll
      for (int fn = 0; fn < 2; ++fn)
        acc[fm][fn] = __builtin_amdgcn_wmma_f32_16x16x32_bf16(
            false, a[fm].v, false, b[fn].v, (short)0, acc[fm][fn], false, false);
    __syncthreads();
  }

  // ---- store ----
#pragma unroll
  for (int fm = 0; fm < 2; ++fm)
#pragma unroll
    for (int fn = 0; fn < 2; ++fn)
#pragma unroll
      for (int r = 0; r < 8; ++r) {
        const int m = mbase + wm * 32 + fm * 16 + r + 8 * hi;
        const int n = nbase + wn * 32 + fn * 16 + lm;
        const float val = acc[fm][fn][r];
        if (OUTMODE == 0) {
          ((float*)Out)[(size_t)m * DMODEL + n] = val;
        } else {
          const int bb = m >> 11, s = m & (SLEN - 1);
          const int hh = n >> 6,  dk = n & (DKH - 1);
          size_t idx;
          if (OUTMODE == 1)
            idx = ((size_t)(bb * NHEAD + hh) * SLEN + s) * DKH + dk;
          else
            idx = ((size_t)(bb * NHEAD + hh) * DKH + dk) * SLEN + s;
          ((unsigned short*)Out)[idx] = f2bf(val);
        }
      }
}

// ---------------------------------------------------------------------------
// Fused attention: one workgroup owns 32 query rows of one (b,h).
//   Phase 1: scores (WMMA, causal-aware) -> 32x2048 fp32 tile in LDS (257KB;
//            only possible thanks to CDNA5's 320KB/WG LDS).
//   Phase 2: row softmax in LDS, single coalesced write of final attn to HBM.
//   Phase 3: ctx = attn @ V with attn from LDS (fp32->bf16 in-register),
//            contraction split across 8 waves, k-loop clipped at the diagonal,
//            partial reduction through reused LDS. ctx stored bf16 [B,S,D].
// ---------------------------------------------------------------------------
__global__ __launch_bounds__(256)
void attention_fused_kernel(const unsigned short* __restrict__ Qb,
                            const unsigned short* __restrict__ Kb,
                            const unsigned short* __restrict__ Vt,
                            float* __restrict__ attn,
                            unsigned short* __restrict__ ctx) {
  __shared__ float sc[32 * SROW];            // 262,656 bytes

  const int bh = blockIdx.y;
  const int bb = bh >> 4, hh = bh & 15;
  const int qb = blockIdx.x * 32;

  const unsigned short* Q  = Qb + (size_t)bh * SLEN * DKH;
  const unsigned short* Kt = Kb + (size_t)bh * SLEN * DKH;
  const unsigned short* Vh = Vt + (size_t)bh * DKH * SLEN;
  float* Ag = attn + (size_t)bh * SLEN * SLEN;

  const int tid = threadIdx.x, l = tid & 31, w = tid >> 5;
  const int lm = l & 15, hi = l >> 4;

  // ---------------- Phase 1: scores strip [32 x 256] per wave ----------------
  FragBF aq[2][2];                           // Q frags: [fm][ks], resident
#pragma unroll
  for (int fm = 0; fm < 2; ++fm)
#pragma unroll
    for (int ks = 0; ks < 2; ++ks) {
      const unsigned short* rp = Q + (qb + fm * 16 + lm) * DKH + ks * 32;
      aq[fm][ks].q[0] = *(const uint4*)(rp + hi * 8);
      aq[fm][ks].q[1] = *(const uint4*)(rp + hi * 8 + 16);
    }

  const int kcbase = w * 256;
  for (int fn = 0; fn < 16; ++fn) {
    const int kc = kcbase + fn * 16;         // 16 key columns
    if (kc > qb + 31) {                      // fully masked chunk
#pragma unroll
      for (int fm = 0; fm < 2; ++fm)
#pragma unroll
        for (int r = 0; r < 8; ++r) {
          const int m = fm * 16 + r + 8 * hi;
          sc[m * SROW + kc + lm] = -1e9f;
        }
      continue;
    }
    v8f acc[2] = {};
#pragma unroll
    for (int ks = 0; ks < 2; ++ks) {
      FragBF bk;
      const unsigned short* rp = Kt + (size_t)(kc + lm) * DKH + ks * 32 + hi * 8;
      bk.q[0] = ((const uint4*)rp)[0];
      bk.q[1] = ((const uint4*)rp)[2];
#pragma unroll
      for (int fm = 0; fm < 2; ++fm)
        acc[fm] = __builtin_amdgcn_wmma_f32_16x16x32_bf16(
            false, aq[fm][ks].v, false, bk.v, (short)0, acc[fm], false, false);
    }
    __builtin_prefetch((const char*)(Kt + (size_t)(kc + 16 + lm) * DKH), 0, 1);
#pragma unroll
    for (int fm = 0; fm < 2; ++fm)
#pragma unroll
      for (int r = 0; r < 8; ++r) {
        const int m = fm * 16 + r + 8 * hi;
        const int k = kc + lm;
        const float v = acc[fm][r] * 0.125f;             // 1/sqrt(64)
        sc[m * SROW + k] = (k > qb + m) ? -1e9f : v;
      }
  }
  __syncthreads();

  // ---------------- Phase 2: softmax, write attn once ----------------
  {
    const int row = tid >> 3;                // 0..31
    const int sub = tid & 7;                 // 0..7 (8 threads per row)
    float* rp = &sc[row * SROW];
    float* gp = Ag + (size_t)(qb + row) * SLEN;

    float m = -3.4e38f;
#pragma unroll 8
    for (int i = 0; i < 64; ++i) {
      float4 v = *(const float4*)&rp[(i * 8 + sub) * 4];
      m = fmaxf(m, fmaxf(fmaxf(v.x, v.y), fmaxf(v.z, v.w)));
    }
    m = fmaxf(m, __shfl_xor(m, 1, 32));
    m = fmaxf(m, __shfl_xor(m, 2, 32));
    m = fmaxf(m, __shfl_xor(m, 4, 32));

    float s = 0.f;
#pragma unroll 8
    for (int i = 0; i < 64; ++i) {
      float4 v = *(const float4*)&rp[(i * 8 + sub) * 4];
      v.x = __expf(v.x - m); v.y = __expf(v.y - m);
      v.z = __expf(v.z - m); v.w = __expf(v.w - m);
      s += v.x + v.y + v.z + v.w;
      *(float4*)&rp[(i * 8 + sub) * 4] = v;
    }
    s += __shfl_xor(s, 1, 32);
    s += __shfl_xor(s, 2, 32);
    s += __shfl_xor(s, 4, 32);
    const float inv = 1.0f / s;

#pragma unroll 8
    for (int i = 0; i < 64; ++i) {
      float4 v = *(const float4*)&rp[(i * 8 + sub) * 4];
      v.x *= inv; v.y *= inv; v.z *= inv; v.w *= inv;
      *(float4*)&rp[(i * 8 + sub) * 4] = v;
      *(float4*)&gp[(i * 8 + sub) * 4] = v;              // final attn to HBM
    }
  }
  __syncthreads();

  // ---------------- Phase 3: ctx partials over this wave's k-slice ----------
  v8f cacc[2][4] = {};
  const int kmax    = qb + 32;               // exclusive (causal)
  const int k_begin = w * 256;
  const int k_end   = (k_begin + 256 < kmax) ? (k_begin + 256) : kmax;

  for (int k0 = k_begin; k0 < k_end; k0 += 32) {
    FragBF a[2];
#pragma unroll
    for (int fm = 0; fm < 2; ++fm) {
      const float* rp = &sc[(fm * 16 + lm) * SROW + k0 + hi * 8];
      float4 x0 = ((const float4*)rp)[0];
      float4 x1 = ((const float4*)rp)[1];
      float4 x2 = ((const float4*)(rp + 16))[0];
      float4 x3 = ((const float4*)(rp + 16))[1];
      a[fm].u[0] = pack2(x0.x, x0.y); a[fm].u[1] = pack2(x0.z, x0.w);
      a[fm].u[2] = pack2(x1.x, x1.y); a[fm].u[3] = pack2(x1.z, x1.w);
      a[fm].u[4] = pack2(x2.x, x2.y); a[fm].u[5] = pack2(x2.z, x2.w);
      a[fm].u[6] = pack2(x3.x, x3.y); a[fm].u[7] = pack2(x3.z, x3.w);
    }
#pragma unroll
    for (int fn = 0; fn < 4; ++fn) {
      FragBF b;
      const unsigned short* vp = Vh + (size_t)(fn * 16 + lm) * SLEN + k0 + hi * 8;
      b.q[0] = ((const uint4*)vp)[0];
      b.q[1] = ((const uint4*)vp)[2];
      __builtin_prefetch((const char*)(vp + 32), 0, 1);  // next k-step
#pragma unroll
      for (int fm = 0; fm < 2; ++fm)
        cacc[fm][fn] = __builtin_amdgcn_wmma_f32_16x16x32_bf16(
            false, a[fm].v, false, b.v, (short)0, cacc[fm][fn], false, false);
    }
  }
  __syncthreads();                           // score tile no longer needed

  // partial reduction through reused LDS (8 waves x 32 x 64 fp32 = 64KB)
  float* pf = sc;
#pragma unroll
  for (int fm = 0; fm < 2; ++fm)
#pragma unroll
    for (int fn = 0; fn < 4; ++fn)
#pragma unroll
      for (int r = 0; r < 8; ++r) {
        const int m = fm * 16 + r + 8 * hi;
        const int n = fn * 16 + lm;
        pf[w * 2048 + m * 64 + n] = cacc[fm][fn][r];
      }
  __syncthreads();

#pragma unroll
  for (int j = 0; j < 8; ++j) {
    const int e = j * 256 + tid;             // 0..2047
    const int m = e >> 6, n = e & 63;
    float sum = 0.f;
#pragma unroll
    for (int ww = 0; ww < 8; ++ww) sum += pf[ww * 2048 + e];
    ctx[((size_t)(bb * SLEN + qb + m)) * DMODEL + hh * DKH + n] = f2bf(sum);
  }
}

// ---------------------------------------------------------------------------
extern "C" void kernel_launch(void* const* d_in, const int* in_sizes, int n_in,
                              void* d_out, int out_size, void* d_ws, size_t ws_size,
                              hipStream_t stream) {
  const float* query = (const float*)d_in[0];
  const float* key   = (const float*)d_in[1];
  const float* value = (const float*)d_in[2];
  // d_in[3]: causal mask — applied analytically (k > q) instead of loading.
  const float* Wq = (const float*)d_in[4];
  const float* Wk = (const float*)d_in[5];
  const float* Wv = (const float*)d_in[6];
  const float* Wo = (const float*)d_in[7];

  // Workspace layout (64 MB total):
  unsigned char* ws = (unsigned char*)d_ws;
  unsigned short* Qb  = (unsigned short*)(ws + (size_t)0);
  unsigned short* Kb  = (unsigned short*)(ws + (size_t)16 * 1024 * 1024);
  unsigned short* Vt  = (unsigned short*)(ws + (size_t)32 * 1024 * 1024);
  unsigned short* ctx = (unsigned short*)(ws + (size_t)48 * 1024 * 1024);

  float* out  = (float*)d_out;
  float* attn = out + (size_t)BATCH * SLEN * DMODEL;

  const dim3 gemmGrid(DMODEL / 64, (BATCH * SLEN) / 128, 1);
  gemm_kernel<false, 1><<<gemmGrid, 256, 0, stream>>>(query, Wq, Qb);
  gemm_kernel<false, 1><<<gemmGrid, 256, 0, stream>>>(key,   Wk, Kb);
  gemm_kernel<false, 2><<<gemmGrid, 256, 0, stream>>>(value, Wv, Vt);

  attention_fused_kernel<<<dim3(SLEN / 32, BATCH * NHEAD), 256, 0, stream>>>(
      Qb, Kb, Vt, attn, ctx);

  gemm_kernel<true, 0><<<gemmGrid, 256, 0, stream>>>(ctx, Wo, out);
}
